// LFADSNET_42889543418152
// MI455X (gfx1250) — compile-verified, matrix operand order checked by reading, not run on hardware
//
#include <hip/hip_runtime.h>
#include <hip/hip_bf16.h>
#include <stdint.h>

// ---------------------------------------------------------------------------
// LFADS forward on gfx1250: bf16 WMMA GEMMs (f32 accumulate) + pointwise GRU.
// Weights converted to bf16 once per call; activations stay fp32 (L2-resident).
// Block tile 64(M) x 128(N), 4 wave32s; each wave computes 32x64 via 2x4
// accumulators -> 8 v_wmma per K-step of 32, halving redundant fragment loads.
// ---------------------------------------------------------------------------

typedef __attribute__((ext_vector_type(16))) __bf16 v16bf;
typedef __attribute__((ext_vector_type(8)))  __bf16 bf16x8;
typedef __attribute__((ext_vector_type(8)))  float  v8f;
typedef __attribute__((ext_vector_type(8)))  float  f32x8;

#define TM 64
#define TN 128

// ---- fragment loaders -----------------------------------------------------
// A (16x32, 16-bit): lanes 0-15 -> M=lane, K in {k0..k0+7} U {k0+16..k0+23};
// lanes 16-31 -> M=lane-16, K in {k0+8..15} U {k0+24..31}. Base offset
// k0 + hi*8 covers both lane halves; second 8 elems are +16 further.
__device__ __forceinline__ v16bf load_a_frag(const float* __restrict__ A, long lda,
                                             int row, int k0, int lane) {
  int sub = lane & 15, hi = lane >> 4;
  const float* p = A + (long)(row + sub) * lda + (k0 + hi * 8);
  f32x8 lo = *(const f32x8*)p;
  f32x8 hh = *(const f32x8*)(p + 16);
  v16bf f;
#pragma unroll
  for (int i = 0; i < 8; ++i) { f[i] = (__bf16)lo[i]; f[8 + i] = (__bf16)hh[i]; }
  return f;
}

// B operand = rows of W [N x K] (since C = A * W^T, column n of B is row n of W).
__device__ __forceinline__ v16bf load_b_frag(const __bf16* __restrict__ W, long ldw,
                                             int col, int k0, int lane) {
  int sub = lane & 15, hi = lane >> 4;
  const __bf16* p = W + (long)(col + sub) * ldw + (k0 + hi * 8);
  bf16x8 lo = *(const bf16x8*)p;
  bf16x8 hh = *(const bf16x8*)(p + 16);
  return __builtin_shufflevector(lo, hh, 0, 1, 2, 3, 4, 5, 6, 7,
                                 8, 9, 10, 11, 12, 13, 14, 15);
}

// ---- generic GEMM: C = act(A[MxK] * W[NxK]^T + bias), dual strided outputs --
__global__ __launch_bounds__(128)
void wmma_gemm(const float* __restrict__ A, long lda,
               const __bf16* __restrict__ W, long ldw,
               const float* __restrict__ bias,
               float* __restrict__ C0, long ldc0,
               float* __restrict__ C1, long ldc1,
               int M, int N, int K, int act) {
  const int lane = threadIdx.x & 31;
  const int wave = threadIdx.x >> 5;      // 0..3
  const int wm = wave & 1;                // 2 waves along M
  const int wn = wave >> 1;               // 2 waves along N
  const int bm = blockIdx.x * TM + wm * 32;
  const int bn = blockIdx.y * TN + wn * 64;

  v8f acc[2][4] = {};

  for (int k0 = 0; k0 < K; k0 += 32) {
    if (k0 + 32 < K) {  // warm next weight K-slab (global_prefetch_b8 path)
      __builtin_prefetch(W + (long)(bn + (lane & 15)) * ldw + (k0 + 32), 0, 1);
    }
    v16bf a0 = load_a_frag(A, lda, bm, k0, lane);
    v16bf a1 = load_a_frag(A, lda, bm + 16, k0, lane);
    v16bf b[4];
#pragma unroll
    for (int ni = 0; ni < 4; ++ni) b[ni] = load_b_frag(W, ldw, bn + 16 * ni, k0, lane);
#pragma unroll
    for (int ni = 0; ni < 4; ++ni) {
      acc[0][ni] = __builtin_amdgcn_wmma_f32_16x16x32_bf16(false, a0, false, b[ni],
                                                           (short)0, acc[0][ni], false, false);
      acc[1][ni] = __builtin_amdgcn_wmma_f32_16x16x32_bf16(false, a1, false, b[ni],
                                                           (short)0, acc[1][ni], false, false);
    }
  }

  // C/D layout: VGPR v -> row M=v (lanes 0-15) / M=v+8 (lanes 16-31), col=lane&15
  const int sub = lane & 15, hi = lane >> 4;
#pragma unroll
  for (int ni = 0; ni < 4; ++ni) {
    const int col = bn + ni * 16 + sub;
    const float bv = bias ? bias[col] : 0.0f;
#pragma unroll
    for (int mi = 0; mi < 2; ++mi) {
#pragma unroll
      for (int v = 0; v < 8; ++v) {
        int row = bm + mi * 16 + v + (hi ? 8 : 0);
        float x = acc[mi][ni][v] + bv;
        if (act == 1) x = fmaxf(x, 0.0f);
        if (C0) C0[(long)row * ldc0 + col] = x;
        if (C1) C1[(long)row * ldc1 + col] = x;
      }
    }
  }
}

// ---- GRU pointwise: gates + state update (+ clip) -------------------------
__device__ __forceinline__ float sigmoidf(float x) { return 1.0f / (1.0f + __expf(-x)); }

__global__ void gru_pointwise(const float* __restrict__ gi, const float* __restrict__ gh,
                              float* __restrict__ h, long hstride,
                              int Bn, int H, int clipmode) {
  int idx = blockIdx.x * blockDim.x + threadIdx.x;
  if (idx >= Bn * H) return;
  int b = idx / H, j = idx - b * H;
  const float* gib = gi + (long)b * 3 * H;
  const float* ghb = gh + (long)b * 3 * H;
  float r = sigmoidf(gib[j] + ghb[j]);
  float z = sigmoidf(gib[H + j] + ghb[H + j]);
  float n = tanhf(gib[2 * H + j] + r * ghb[2 * H + j]);
  float* hp = h + (long)b * hstride + j;
  float hv = *hp;
  float hn = (1.0f - z) * n + z * hv;
  if (clipmode == 1) hn = fminf(hn, 5.0f);                      // encoder: min only
  else if (clipmode == 2) hn = fminf(fmaxf(hn, -5.0f), 5.0f);   // generator: clamp
  *hp = hn;
}

// ---- reparameterize + KL --------------------------------------------------
__global__ void reparam_kl(const float* __restrict__ mean, const float* __restrict__ logvar,
                           const float* __restrict__ eps,
                           const float* __restrict__ mu_p, const float* __restrict__ lk_p,
                           float* __restrict__ g, float* __restrict__ kl_out,
                           int total, float invB) {
  __shared__ float red[256];
  int idx = blockIdx.x * 256 + threadIdx.x;
  float acc = 0.0f;
  if (idx < total) {
    float lv = fmaxf(logvar[idx], -9.210340371976182f);   // log(1e-4)
    float m = mean[idx];
    g[idx] = eps[idx] * __expf(0.5f * lv) + m;
    float lk = *lk_p;
    float d = m - *mu_p;
    acc = 0.5f * (lk - lv + (__expf(lv) + d * d) * __expf(-lk) - 1.0f) * invB;
  }
  red[threadIdx.x] = acc;
  __syncthreads();
  for (int s = 128; s > 0; s >>= 1) {
    if ((int)threadIdx.x < s) red[threadIdx.x] += red[threadIdx.x + s];
    __syncthreads();
  }
  if (threadIdx.x == 0) atomicAdd(kl_out, red[0]);
}

// ---- utilities ------------------------------------------------------------
__global__ void f32_to_bf16(const float* __restrict__ in, __bf16* __restrict__ out, long n) {
  long i = (long)blockIdx.x * blockDim.x + threadIdx.x;
  if (i < n) out[i] = (__bf16)in[i];
}
__global__ void zero_f32(float* __restrict__ p, long n) {
  long i = (long)blockIdx.x * blockDim.x + threadIdx.x;
  if (i < n) p[i] = 0.0f;
}

// ---------------------------------------------------------------------------
extern "C" void kernel_launch(void* const* d_in, const int* in_sizes, int n_in,
                              void* d_out, int out_size, void* d_ws, size_t ws_size,
                              hipStream_t stream) {
  (void)in_sizes; (void)n_in; (void)out_size; (void)ws_size;
  constexpr int Bn = 512, T = 400, DIN = 256, ENC = 512, G = 512, FAC = 128;

  const float* x      = (const float*)d_in[0];
  const float* eps_g0 = (const float*)d_in[1];
  const float* Wih_f  = (const float*)d_in[2];   const float* Whh_f = (const float*)d_in[3];
  const float* bih_f  = (const float*)d_in[4];   const float* bhh_f = (const float*)d_in[5];
  const float* Wih_b  = (const float*)d_in[6];   const float* Whh_b = (const float*)d_in[7];
  const float* bih_b  = (const float*)d_in[8];   const float* bhh_b = (const float*)d_in[9];
  const float* Wih_g  = (const float*)d_in[10];  const float* Whh_g = (const float*)d_in[11];
  const float* bih_g  = (const float*)d_in[12];  const float* bhh_g = (const float*)d_in[13];
  const float* Wm = (const float*)d_in[14];  const float* bm = (const float*)d_in[15];
  const float* Wv = (const float*)d_in[16];  const float* bv = (const float*)d_in[17];
  const float* Wf = (const float*)d_in[18];  const float* bf = (const float*)d_in[19];
  const float* W1 = (const float*)d_in[20];  const float* b1 = (const float*)d_in[21];
  const float* W2 = (const float*)d_in[22];  const float* b2 = (const float*)d_in[23];
  const float* W3 = (const float*)d_in[24];  const float* b3 = (const float*)d_in[25];
  const float* mu_p = (const float*)d_in[26];
  const float* lk_p = (const float*)d_in[27];

  float* out = (float*)d_out;
  const long FAC_OFF  = (long)Bn * T * DIN;                // 52,428,800
  const long KL_OFF   = FAC_OFF + (long)Bn * T * FAC;      // 78,643,200
  float* out_pred = out;
  float* out_fac  = out + FAC_OFF;
  float* out_kl   = out + KL_OFF;

  // ---- workspace carve-up (deterministic each call) ----
  char* ws = (char*)d_ws;
  size_t off = 0;
  auto carve = [&](size_t bytes) { size_t o = off; off = (off + bytes + 255) & ~(size_t)255; return o; };
  auto bw = [&](long n) { return (__bf16*)(ws + carve((size_t)n * 2)); };
  auto fw = [&](long n) { return (float*)(ws + carve((size_t)n * 4)); };

  __bf16* wihf = bw((long)3 * ENC * DIN);  __bf16* whhf = bw((long)3 * ENC * ENC);
  __bf16* wihb = bw((long)3 * ENC * DIN);  __bf16* whhb = bw((long)3 * ENC * ENC);
  __bf16* wihg = bw((long)3 * G * FAC);    __bf16* whhg = bw((long)3 * G * G);
  __bf16* wm   = bw((long)G * 2 * ENC);    __bf16* wv   = bw((long)G * 2 * ENC);
  __bf16* wf   = bw((long)FAC * G);
  __bf16* w1   = bw((long)DIN * FAC);      __bf16* w2 = bw((long)DIN * DIN);
  __bf16* w3   = bw((long)DIN * DIN);

  float* gi   = fw((long)Bn * 3 * ENC);
  float* gh   = fw((long)Bn * 3 * ENC);
  float* egen = fw((long)Bn * 2 * ENC);   // [ef | eb] packed per row, stride 1024
  float* mean = fw((long)Bn * G);
  float* lvar = fw((long)Bn * G);
  float* g    = fw((long)Bn * G);
  float* f    = fw((long)Bn * FAC);
  float* o1   = fw((long)Bn * DIN);
  float* o2   = fw((long)Bn * DIN);

  auto cvt = [&](const float* src, __bf16* dst, long n) {
    f32_to_bf16<<<(unsigned)((n + 255) / 256), 256, 0, stream>>>(src, dst, n);
  };
  cvt(Wih_f, wihf, (long)3 * ENC * DIN);  cvt(Whh_f, whhf, (long)3 * ENC * ENC);
  cvt(Wih_b, wihb, (long)3 * ENC * DIN);  cvt(Whh_b, whhb, (long)3 * ENC * ENC);
  cvt(Wih_g, wihg, (long)3 * G * FAC);    cvt(Whh_g, whhg, (long)3 * G * G);
  cvt(Wm, wm, (long)G * 2 * ENC);         cvt(Wv, wv, (long)G * 2 * ENC);
  cvt(Wf, wf, (long)FAC * G);
  cvt(W1, w1, (long)DIN * FAC);           cvt(W2, w2, (long)DIN * DIN);
  cvt(W3, w3, (long)DIN * DIN);

  zero_f32<<<(Bn * 2 * ENC + 255) / 256, 256, 0, stream>>>(egen, (long)Bn * 2 * ENC);
  zero_f32<<<1, 256, 0, stream>>>(out_kl, 1);

  auto gemm = [&](const float* A, long lda, const __bf16* W, long ldw, const float* bias_,
                  float* C0, long ldc0, float* C1, long ldc1, int N, int K, int act) {
    dim3 grid(Bn / TM, N / TN);
    wmma_gemm<<<grid, 128, 0, stream>>>(A, lda, W, ldw, bias_, C0, ldc0, C1, ldc1, Bn, N, K, act);
  };
  auto gru = [&](float* h, long hs, int H, int clip) {
    gru_pointwise<<<(Bn * H + 255) / 256, 256, 0, stream>>>(gi, gh, h, hs, Bn, H, clip);
  };

  // ---- bidirectional encoder: T serial steps ----
  for (int s = 0; s < T; ++s) {
    // forward dir: x_s slice as strided A (lda = T*DIN)
    gemm(x + (long)s * DIN, (long)T * DIN, wihf, DIN, bih_f, gi, 3 * ENC, nullptr, 0, 3 * ENC, DIN, 0);
    gemm(egen, 2 * ENC, whhf, ENC, bhh_f, gh, 3 * ENC, nullptr, 0, 3 * ENC, ENC, 0);
    gru(egen, 2 * ENC, ENC, 1);
    // backward dir: x_{T-1-s}
    gemm(x + (long)(T - 1 - s) * DIN, (long)T * DIN, wihb, DIN, bih_b, gi, 3 * ENC, nullptr, 0, 3 * ENC, DIN, 0);
    gemm(egen + ENC, 2 * ENC, whhb, ENC, bhh_b, gh, 3 * ENC, nullptr, 0, 3 * ENC, ENC, 0);
    gru(egen + ENC, 2 * ENC, ENC, 1);
  }

  // ---- g0 posterior + reparameterize + KL ----
  gemm(egen, 2 * ENC, wm, 2 * ENC, bm, mean, G, nullptr, 0, G, 2 * ENC, 0);
  gemm(egen, 2 * ENC, wv, 2 * ENC, bv, lvar, G, nullptr, 0, G, 2 * ENC, 0);
  reparam_kl<<<(Bn * G + 255) / 256, 256, 0, stream>>>(mean, lvar, eps_g0, mu_p, lk_p,
                                                       g, out_kl, Bn * G, 1.0f / Bn);
  gemm(g, G, wf, G, bf, f, FAC, nullptr, 0, FAC, G, 0);   // initial factors (not recorded)

  // ---- generator: T serial steps ----
  for (int t = 0; t < T; ++t) {
    gemm(f, FAC, wihg, FAC, bih_g, gi, 3 * G, nullptr, 0, 3 * G, FAC, 0);
    gemm(g, G, whhg, G, bhh_g, gh, 3 * G, nullptr, 0, 3 * G, G, 0);
    gru(g, G, G, 2);
    // factors: next-step buffer + strided record into d_out[B,T,FAC]
    gemm(g, G, wf, G, bf, f, FAC, out_fac + (long)t * FAC, (long)T * FAC, FAC, G, 0);
    // reconstruction MLP (relu, relu, linear -> d_out[B,T,DIN])
    gemm(f, FAC, w1, FAC, b1, o1, DIN, nullptr, 0, DIN, FAC, 1);
    gemm(o1, DIN, w2, DIN, b2, o2, DIN, nullptr, 0, DIN, DIN, 1);
    gemm(o2, DIN, w3, DIN, b3, out_pred + (long)t * DIN, (long)T * DIN, nullptr, 0, DIN, DIN, 0);
  }
}
